// StateSpaceModel_64330020159810
// MI455X (gfx1250) — matrix operand order repaired and checked by reading, not compiled, not run on hardware
//
#include <hip/hip_runtime.h>
#include <math.h>

// ---------------- problem constants ----------------
#define TSTEPS 200
#define BATCH  128
#define ADIM   32
#define ZDIM   32
#define KMIX   3
#define HIDN   128
#define BM     16            // batch rows per workgroup
#define NWG    (BATCH / BM)  // 8 workgroups

typedef _Float16 h8   __attribute__((ext_vector_type(8)));
typedef _Float16 h16v __attribute__((ext_vector_type(16)));
typedef float    f32x2 __attribute__((ext_vector_type(2)));
typedef float    f32x8 __attribute__((ext_vector_type(8)));

#if __has_builtin(__builtin_amdgcn_wmma_f32_16x16x4_f32)
#define HAVE_WMMA_F32X4 1
#endif

// ---------------- LDS layout (bytes) ----------------
#define OFF_P   0        // 16 x 32x32 f32 covariances        (65536)
#define OFF_SCR 65536    // 8 waves x 4 x 32x32 f32 scratch   (131072)
#define OFF_R   196608   // R matrix f32                      (4096)
#define OFF_Q   200704   // Q matrix f32                      (4096)
#define OFF_MP  204800   // mean_p 16x32 f32                  (2048)
#define OFF_MT  206848   // mean_t 16x32 f32                  (2048)
#define OFF_WC  208896   // w current 16x4 f32                (256)
#define OFF_WN  209152   // w next    16x4 f32                (256)
#define OFF_LG  209408   // logits    16x4 f32                (256)
#define OFF_VI  209664   // per-wave innov 8x32 f32           (1024)
#define OFF_XH  210688   // x tile 16x32 f16                  (1024)
#define OFF_H1  211712   // h1     16x128 f16                 (4096)
#define OFF_H2  215808   // h2     16x128 f16                 (4096)
#define SMEM_BYTES 219904

struct KP {
    const float *asrc, *AK, *CK, *QL, *RL, *imean, *icov;
    const float *bi1, *bh1, *bi2, *bh2, *Wout, *bout;
    const _Float16 *wih1, *whh1, *wih2, *whh2;
    float *o_means, *o_covs, *o_nmeans, *o_ncovs, *o_mAs, *o_mCs;
};

// ---------------- helpers ----------------
static __device__ inline float sigm(float x) { return 1.0f / (1.0f + expf(-x)); }

static __device__ inline f32x8 wmma_f16(h16v a, h16v b, f32x8 c) {
    return __builtin_amdgcn_wmma_f32_16x16x32_f16(false, a, false, b, (short)0, c, false, false);
}

// A-fragment (16x32 f16, rows = M) from row-major [16][rs] halfs.
// lane L: M = L&15, K(e) = kbase + (e<8?0:16) + (L>>4)*8 + (e&7)
static __device__ inline h16v lda16(const _Float16 *X, int rs, int kbase, int lane) {
    int m = lane & 15, kh = lane >> 4;
    const h8 *p0 = (const h8 *)(X + m * rs + kbase + kh * 8);
    const h8 *p1 = (const h8 *)(X + m * rs + kbase + 16 + kh * 8);
    h8 lo = *p0, hi = *p1;
    return __builtin_shufflevector(lo, hi, 0, 1, 2, 3, 4, 5, 6, 7, 8, 9, 10, 11, 12, 13, 14, 15);
}

// B-fragment (32x16 f16, K x N) from W stored [N][kdim] row-major (i.e. W^T K-major).
// lane L: N = nbase + (L&15), K(e) = kbase + (e<8?0:16) + (L>>4)*8 + (e&7)
static __device__ inline h16v ldb16(const _Float16 *W, int kdim, int nbase, int kbase, int lane) {
    const h8 *p = (const h8 *)(W + (size_t)(nbase + (lane & 15)) * kdim + kbase);
    h8 lo = p[lane >> 4], hi = p[2 + (lane >> 4)];
    return __builtin_shufflevector(lo, hi, 0, 1, 2, 3, 4, 5, 6, 7, 8, 9, 10, 11, 12, 13, 14, 15);
}

// 32x32 f32 matmul in LDS using V_WMMA_F32_16X16X4_F32 (2x2 D tiles, 8 K-steps).
// EPI: 0 -> D = A*B ; 1 -> D = A*B + E ; 2 -> D = E - A*B
template <bool TA, bool TB, int EPI>
static __device__ inline void mm32(float *D, const float *A, const float *B,
                                   const float *E, int lane) {
    const int kh = lane >> 4, l15 = lane & 15;
    for (int dm = 0; dm < 2; ++dm)
        for (int dn = 0; dn < 2; ++dn) {
            const int mb = dm * 16 + l15;
            const int nb = dn * 16 + l15;
#ifdef HAVE_WMMA_F32X4
            f32x8 acc = {0, 0, 0, 0, 0, 0, 0, 0};
            for (int kt = 0; kt < 8; ++kt) {
                int kk = kt * 4 + kh * 2;
                f32x2 av, bv;
                av[0] = TA ? A[kk * 32 + mb] : A[mb * 32 + kk];
                av[1] = TA ? A[(kk + 1) * 32 + mb] : A[mb * 32 + kk + 1];
                bv[0] = TB ? B[nb * 32 + kk] : B[kk * 32 + nb];
                bv[1] = TB ? B[nb * 32 + kk + 1] : B[kk * 32 + nb + 1];
                acc = __builtin_amdgcn_wmma_f32_16x16x4_f32(false, av, false, bv,
                                                            (short)0, acc, false, false);
            }
#pragma unroll
            for (int v = 0; v < 8; ++v) {
                int M = dm * 16 + kh * 8 + v, N = nb;
                float r = acc[v];
                if (EPI == 1) r += E[M * 32 + N];
                else if (EPI == 2) r = E[M * 32 + N] - r;
                D[M * 32 + N] = r;
            }
#else
#pragma unroll
            for (int v = 0; v < 8; ++v) {
                int M = dm * 16 + kh * 8 + v, N = nb;
                float r = 0.f;
                for (int k = 0; k < 32; ++k)
                    r += (TA ? A[k * 32 + M] : A[M * 32 + k]) *
                         (TB ? B[N * 32 + k] : B[k * 32 + N]);
                if (EPI == 1) r += E[M * 32 + N];
                else if (EPI == 2) r = E[M * 32 + N] - r;
                D[M * 32 + N] = r;
            }
#endif
        }
}

// In-wave Gauss-Jordan: S (SPD 32x32) , Y <- S^-1 * Y. lane = column.
static __device__ inline void gj_solve(float *S, float *Y, int lane) {
    for (int i = 0; i < 32; ++i) {
        float rp = 1.0f / S[i * 32 + i];
        float spj = S[i * 32 + lane] * rp;
        float ypj = Y[i * 32 + lane] * rp;
        S[i * 32 + lane] = spj;
        Y[i * 32 + lane] = ypj;
        for (int r = 0; r < 32; ++r) {
            if (r == i) continue;
            float f = S[r * 32 + i];           // broadcast read (before lane i stores)
            S[r * 32 + lane] -= f * spj;
            Y[r * 32 + lane] -= f * ypj;
        }
    }
}

static __device__ inline void sym32(float *M, int lane) {
    for (int r = 0; r < lane; ++r) {
        float x = 0.5f * (M[r * 32 + lane] + M[lane * 32 + r]);
        M[r * 32 + lane] = x;
        M[lane * 32 + r] = x;
    }
}

static __device__ inline void store32(float *g, const float *M, int lane) {
    for (int r = 0; r < 32; ++r) g[r * 32 + lane] = M[r * 32 + lane];
}

// ---------------- prep kernels ----------------
__global__ void cvt_f32_f16(const float *s, _Float16 *d, int n) {
    for (int i = blockIdx.x * blockDim.x + threadIdx.x; i < n; i += gridDim.x * blockDim.x)
        d[i] = (_Float16)s[i];
}
__global__ void copy_f32(const float *s, float *d, int n) {
    for (int i = blockIdx.x * blockDim.x + threadIdx.x; i < n; i += gridDim.x * blockDim.x)
        d[i] = s[i];
}

// ---------------- main persistent scan kernel ----------------
__global__ __launch_bounds__(256) void ssm_scan(KP p) {
    extern __shared__ char sm[];
    float    *Pm    = (float *)(sm + OFF_P);
    float    *scr   = (float *)(sm + OFF_SCR);
    float    *Rm    = (float *)(sm + OFF_R);
    float    *Qm    = (float *)(sm + OFF_Q);
    float    *meanP = (float *)(sm + OFF_MP);
    float    *meanT = (float *)(sm + OFF_MT);
    float    *wcur  = (float *)(sm + OFF_WC);
    float    *wnxt  = (float *)(sm + OFF_WN);
    float    *logitS= (float *)(sm + OFF_LG);
    float    *vecI  = (float *)(sm + OFF_VI);
    _Float16 *xh    = (_Float16 *)(sm + OFF_XH);
    _Float16 *h1h   = (_Float16 *)(sm + OFF_H1);
    _Float16 *h2h   = (_Float16 *)(sm + OFF_H2);

    const int tid = threadIdx.x;
    const int wv = tid >> 5, lane = tid & 31;
    const int kh = lane >> 4, l15 = lane & 15;
    const int hs = wv * 16;          // this wave's hidden slice base
    const int hcol = hs + l15;       // hidden column owned by this lane
    const int bg0 = blockIdx.x * BM; // global batch base for this WG

    // ---- init Q = Lq Lq^T + 1e-3 I ; R = Lr Lr^T + 1e-3 I ----
    for (int e = tid; e < 1024; e += 256) {
        int r = e >> 5, c = e & 31;
        float sq = 0.f, sr = 0.f;
        for (int j = 0; j < 32; ++j) {
            sq += p.QL[r * 32 + j] * p.QL[c * 32 + j];
            sr += p.RL[r * 32 + j] * p.RL[c * 32 + j];
        }
        Qm[e] = sq + ((r == c) ? 1e-3f : 0.f);
        Rm[e] = sr + ((r == c) ? 1e-3f : 0.f);
    }
    // ---- init state ----
    for (int e = tid; e < BM * 1024; e += 256) Pm[e] = p.icov[e & 1023];
    for (int e = tid; e < BM * 32; e += 256) meanP[e] = p.imean[e & 31];
    for (int e = tid; e < BM * HIDN; e += 256) { h1h[e] = (_Float16)0.f; h2h[e] = (_Float16)0.f; }
    if (tid < BM) {
        wcur[tid * 4 + 0] = 1.0f / 3.0f; wcur[tid * 4 + 1] = 1.0f / 3.0f;
        wcur[tid * 4 + 2] = 1.0f / 3.0f; wcur[tid * 4 + 3] = 0.f;
    }
    // mat_As[0] = mean of the K transition matrices (w0 uniform)
    for (int e = tid; e < 1024; e += 256) {
        float a0 = (p.AK[e] + p.AK[1024 + e] + p.AK[2048 + e]) * (1.0f / 3.0f);
        for (int b = 0; b < BM; ++b) p.o_mAs[(size_t)(bg0 + b) * 1024 + e] = a0;
    }
    // LSTM biases are step-invariant: hoist
    const float bI1 = p.bi1[hcol] + p.bh1[hcol];
    const float bF1 = p.bi1[128 + hcol] + p.bh1[128 + hcol];
    const float bG1 = p.bi1[256 + hcol] + p.bh1[256 + hcol];
    const float bO1 = p.bi1[384 + hcol] + p.bh1[384 + hcol];
    const float bI2 = p.bi2[hcol] + p.bh2[hcol];
    const float bF2 = p.bi2[128 + hcol] + p.bh2[128 + hcol];
    const float bG2 = p.bi2[256 + hcol] + p.bh2[256 + hcol];
    const float bO2 = p.bi2[384 + hcol] + p.bh2[384 + hcol];

    float c1[8], c2[8];
#pragma unroll
    for (int v = 0; v < 8; ++v) { c1[v] = 0.f; c2[v] = 0.f; }

    __syncthreads();

    for (int t = 0; t < TSTEPS; ++t) {
        // ---- stage a_t slice as f16 ----
        for (int e = tid; e < BM * ADIM; e += 256) {
            int b = e >> 5, r = e & 31;
            xh[b * 32 + r] = (_Float16)p.asrc[((size_t)t * BATCH + bg0 + b) * ADIM + r];
        }
        if (t + 1 < TSTEPS && tid < BM)
            __builtin_prefetch(&p.asrc[((size_t)(t + 1) * BATCH + bg0 + tid) * ADIM], 0, 1);
        __syncthreads();

        // ---- LSTM layer 1: gates[16 x 512] via f16 WMMA, wave owns hidden slice hs ----
        float h1n[8];
        {
            f32x8 gi = {0,0,0,0,0,0,0,0}, gf = {0,0,0,0,0,0,0,0};
            f32x8 gg = {0,0,0,0,0,0,0,0}, go = {0,0,0,0,0,0,0,0};
            h16v ax = lda16(xh, 32, 0, lane);
            gi = wmma_f16(ax, ldb16(p.wih1, 32, 0   + hs, 0, lane), gi);
            gf = wmma_f16(ax, ldb16(p.wih1, 32, 128 + hs, 0, lane), gf);
            gg = wmma_f16(ax, ldb16(p.wih1, 32, 256 + hs, 0, lane), gg);
            go = wmma_f16(ax, ldb16(p.wih1, 32, 384 + hs, 0, lane), go);
            for (int kt = 0; kt < 4; ++kt) {
                h16v ah = lda16(h1h, HIDN, kt * 32, lane);
                gi = wmma_f16(ah, ldb16(p.whh1, HIDN, 0   + hs, kt * 32, lane), gi);
                gf = wmma_f16(ah, ldb16(p.whh1, HIDN, 128 + hs, kt * 32, lane), gf);
                gg = wmma_f16(ah, ldb16(p.whh1, HIDN, 256 + hs, kt * 32, lane), gg);
                go = wmma_f16(ah, ldb16(p.whh1, HIDN, 384 + hs, kt * 32, lane), go);
            }
#pragma unroll
            for (int v = 0; v < 8; ++v) {
                float iv = sigm(gi[v] + bI1), fv = sigm(gf[v] + bF1);
                float gv = tanhf(gg[v] + bG1), ov = sigm(go[v] + bO1);
                float cv = fv * c1[v] + iv * gv;
                c1[v] = cv;
                h1n[v] = ov * tanhf(cv);
            }
        }
        __syncthreads();
#pragma unroll
        for (int v = 0; v < 8; ++v) h1h[(kh * 8 + v) * HIDN + hcol] = (_Float16)h1n[v];
        __syncthreads();

        // ---- LSTM layer 2 ----
        float h2n[8];
        {
            f32x8 gi = {0,0,0,0,0,0,0,0}, gf = {0,0,0,0,0,0,0,0};
            f32x8 gg = {0,0,0,0,0,0,0,0}, go = {0,0,0,0,0,0,0,0};
            for (int kt = 0; kt < 4; ++kt) {
                h16v ah = lda16(h1h, HIDN, kt * 32, lane);
                gi = wmma_f16(ah, ldb16(p.wih2, HIDN, 0   + hs, kt * 32, lane), gi);
                gf = wmma_f16(ah, ldb16(p.wih2, HIDN, 128 + hs, kt * 32, lane), gf);
                gg = wmma_f16(ah, ldb16(p.wih2, HIDN, 256 + hs, kt * 32, lane), gg);
                go = wmma_f16(ah, ldb16(p.wih2, HIDN, 384 + hs, kt * 32, lane), go);
            }
            for (int kt = 0; kt < 4; ++kt) {
                h16v ah = lda16(h2h, HIDN, kt * 32, lane);
                gi = wmma_f16(ah, ldb16(p.whh2, HIDN, 0   + hs, kt * 32, lane), gi);
                gf = wmma_f16(ah, ldb16(p.whh2, HIDN, 128 + hs, kt * 32, lane), gf);
                gg = wmma_f16(ah, ldb16(p.whh2, HIDN, 256 + hs, kt * 32, lane), gg);
                go = wmma_f16(ah, ldb16(p.whh2, HIDN, 384 + hs, kt * 32, lane), go);
            }
#pragma unroll
            for (int v = 0; v < 8; ++v) {
                float iv = sigm(gi[v] + bI2), fv = sigm(gf[v] + bF2);
                float gv = tanhf(gg[v] + bG2), ov = sigm(go[v] + bO2);
                float cv = fv * c2[v] + iv * gv;
                c2[v] = cv;
                h2n[v] = ov * tanhf(cv);
            }
        }
        __syncthreads();
#pragma unroll
        for (int v = 0; v < 8; ++v) h2h[(kh * 8 + v) * HIDN + hcol] = (_Float16)h2n[v];
        __syncthreads();

        // ---- mixture weights: softmax(h2 @ Wout^T + bout) ----
        if (tid < BM * KMIX) {
            int b = tid & 15, k = tid >> 4;
            float s = p.bout[k];
            for (int j = 0; j < HIDN; ++j) s += (float)h2h[b * HIDN + j] * p.Wout[k * HIDN + j];
            logitS[b * 4 + k] = s;
        }
        __syncthreads();
        if (tid < BM) {
            float l0 = logitS[tid * 4], l1 = logitS[tid * 4 + 1], l2 = logitS[tid * 4 + 2];
            float m = fmaxf(l0, fmaxf(l1, l2));
            float e0 = expf(l0 - m), e1 = expf(l1 - m), e2 = expf(l2 - m);
            float inv = 1.0f / (e0 + e1 + e2);
            wnxt[tid * 4] = e0 * inv; wnxt[tid * 4 + 1] = e1 * inv; wnxt[tid * 4 + 2] = e2 * inv;
        }
        __syncthreads();

        // ---- Kalman update: each wave handles 2 batch elements with f32 WMMA ----
        float *Cb = scr + wv * 4096, *CPb = Cb + 1024, *Sb = CPb + 1024, *Yb = Sb + 1024;
        for (int s = 0; s < 2; ++s) {
            const int bb = wv * 2 + s, bgl = bg0 + bb;
            float *Pb = Pm + bb * 1024;
            // matC = sum_k w[k] C_K[k]  -> Cb ; emit mat_Cs[t]
            {
                float w0 = wcur[bb * 4], w1 = wcur[bb * 4 + 1], w2 = wcur[bb * 4 + 2];
                for (int r = 0; r < 32; ++r)
                    Cb[r * 32 + lane] = w0 * p.CK[r * 32 + lane] +
                                        w1 * p.CK[1024 + r * 32 + lane] +
                                        w2 * p.CK[2048 + r * 32 + lane];
                store32(p.o_mCs + ((size_t)t * BATCH + bgl) * 1024, Cb, lane);
            }
            // CP = C * P ; S = CP * C^T + R
            mm32<false, false, 0>(CPb, Cb, Pb, nullptr, lane);
            mm32<false, true, 1>(Sb, CPb, Cb, Rm, lane);
            // innov = a_t - C * mean_p
            {
                float sv = p.asrc[((size_t)t * BATCH + bgl) * ADIM + lane];
                for (int j = 0; j < 32; ++j) sv -= Cb[lane * 32 + j] * meanP[bb * 32 + j];
                vecI[wv * 32 + lane] = sv;
            }
            // Y = S^-1 * CP (Kt = Y^T)
            for (int r = 0; r < 32; ++r) Yb[r * 32 + lane] = CPb[r * 32 + lane];
            gj_solve(Sb, Yb, lane);
            // mean_t = mean_p + Kt * innov
            {
                float mt = meanP[bb * 32 + lane];
                for (int j = 0; j < 32; ++j) mt += Yb[j * 32 + lane] * vecI[wv * 32 + j];
                meanT[bb * 32 + lane] = mt;
                p.o_means[((size_t)t * BATCH + bgl) * ZDIM + lane] = mt;
            }
            // cov_t = P - Y^T * CP  (into Cb), symmetrize, emit
            mm32<true, false, 2>(Cb, Yb, CPb, Pb, lane);
            sym32(Cb, lane);
            store32(p.o_covs + ((size_t)t * BATCH + bgl) * 1024, Cb, lane);
            // matA_next from w_next (into Sb) ; emit mat_As[t+1]
            {
                float a0 = wnxt[bb * 4], a1 = wnxt[bb * 4 + 1], a2 = wnxt[bb * 4 + 2];
                for (int r = 0; r < 32; ++r)
                    Sb[r * 32 + lane] = a0 * p.AK[r * 32 + lane] +
                                        a1 * p.AK[1024 + r * 32 + lane] +
                                        a2 * p.AK[2048 + r * 32 + lane];
                store32(p.o_mAs + ((size_t)(t + 1) * BATCH + bgl) * 1024, Sb, lane);
            }
            // cov_pn = A * cov_t * A^T + Q (into persistent P), symmetrize, emit
            mm32<false, false, 0>(Yb, Sb, Cb, nullptr, lane);       // AC = A * cov_t
            mm32<false, true, 1>(Pb, Yb, Sb, Qm, lane);             // P' = AC * A^T + Q
            sym32(Pb, lane);
            store32(p.o_ncovs + ((size_t)t * BATCH + bgl) * 1024, Pb, lane);
            // mean_pn = A * mean_t
            {
                float mp = 0.f;
                for (int j = 0; j < 32; ++j) mp += Sb[lane * 32 + j] * meanT[bb * 32 + j];
                meanP[bb * 32 + lane] = mp;
                p.o_nmeans[((size_t)t * BATCH + bgl) * ZDIM + lane] = mp;
            }
        }
        __syncthreads();
        if (tid < BM * 4) wcur[tid] = wnxt[tid];
        __syncthreads();
    }

    // ---- trailing mat_Cs[T] from w_last ----
    {
        float *Cb = scr + wv * 4096;
        for (int s = 0; s < 2; ++s) {
            const int bb = wv * 2 + s, bgl = bg0 + bb;
            float w0 = wcur[bb * 4], w1 = wcur[bb * 4 + 1], w2 = wcur[bb * 4 + 2];
            for (int r = 0; r < 32; ++r)
                Cb[r * 32 + lane] = w0 * p.CK[r * 32 + lane] +
                                    w1 * p.CK[1024 + r * 32 + lane] +
                                    w2 * p.CK[2048 + r * 32 + lane];
            store32(p.o_mCs + ((size_t)TSTEPS * BATCH + bgl) * 1024, Cb, lane);
        }
    }
}

// ---------------- launch ----------------
extern "C" void kernel_launch(void *const *d_in, const int *in_sizes, int n_in,
                              void *d_out, int out_size, void *d_ws, size_t ws_size,
                              hipStream_t stream) {
    (void)in_sizes; (void)n_in; (void)out_size; (void)ws_size;
    const float *as_  = (const float *)d_in[0];
    const float *AK   = (const float *)d_in[1];
    const float *CK   = (const float *)d_in[2];
    const float *QL   = (const float *)d_in[3];
    const float *RL   = (const float *)d_in[4];
    const float *im   = (const float *)d_in[5];
    const float *ic   = (const float *)d_in[6];
    const float *Wih1 = (const float *)d_in[7];
    const float *Whh1 = (const float *)d_in[8];
    const float *bi1  = (const float *)d_in[9];
    const float *bh1  = (const float *)d_in[10];
    const float *Wih2 = (const float *)d_in[11];
    const float *Whh2 = (const float *)d_in[12];
    const float *bi2  = (const float *)d_in[13];
    const float *bh2  = (const float *)d_in[14];
    const float *Wout = (const float *)d_in[15];
    const float *bout = (const float *)d_in[16];

    // f16 weight copies in workspace
    _Float16 *ws = (_Float16 *)d_ws;
    _Float16 *wih1 = ws;                 // 512*32
    _Float16 *whh1 = ws + 16384;         // 512*128
    _Float16 *wih2 = ws + 81920;         // 512*128
    _Float16 *whh2 = ws + 147456;        // 512*128
    cvt_f32_f16<<<64, 256, 0, stream>>>(Wih1, wih1, 16384);
    cvt_f32_f16<<<256, 256, 0, stream>>>(Whh1, whh1, 65536);
    cvt_f32_f16<<<256, 256, 0, stream>>>(Wih2, wih2, 65536);
    cvt_f32_f16<<<256, 256, 0, stream>>>(Whh2, whh2, 65536);

    // output layout (flat float, reference return order)
    float *out = (float *)d_out;
    const size_t N_TBZ = (size_t)TSTEPS * BATCH * ZDIM;          // 819200
    const size_t N_TBZZ = (size_t)TSTEPS * BATCH * ZDIM * ZDIM;  // 26214400
    const size_t N_T1BZZ = (size_t)(TSTEPS + 1) * BATCH * ZDIM * ZDIM;
    float *o_means  = out;
    float *o_covs   = o_means + N_TBZ;
    float *o_nmeans = o_covs + N_TBZZ;
    float *o_ncovs  = o_nmeans + N_TBZ;
    float *o_mAs    = o_ncovs + N_TBZZ;
    float *o_mCs    = o_mAs + N_T1BZZ;
    float *o_asw    = o_mCs + N_T1BZZ;

    // as_for_weight is a straight copy of the observations
    copy_f32<<<1024, 256, 0, stream>>>(as_, o_asw, (int)N_TBZ);

    KP p;
    p.asrc = as_; p.AK = AK; p.CK = CK; p.QL = QL; p.RL = RL; p.imean = im; p.icov = ic;
    p.bi1 = bi1; p.bh1 = bh1; p.bi2 = bi2; p.bh2 = bh2; p.Wout = Wout; p.bout = bout;
    p.wih1 = wih1; p.whh1 = whh1; p.wih2 = wih2; p.whh2 = whh2;
    p.o_means = o_means; p.o_covs = o_covs; p.o_nmeans = o_nmeans;
    p.o_ncovs = o_ncovs; p.o_mAs = o_mAs; p.o_mCs = o_mCs;

    ssm_scan<<<NWG, 256, SMEM_BYTES, stream>>>(p);
}
